// RoAttention_81131932221845
// MI455X (gfx1250) — matrix operand order repaired and checked by reading, not compiled, not run on hardware
//
#include <hip/hip_runtime.h>
#include <hip/hip_bf16.h>

// ---------------------------------------------------------------------------
// Types for CDNA5 WMMA bf16: D(f32 16x16) = A(bf16 16x32) * B(bf16 32x16) + C
// ---------------------------------------------------------------------------
typedef __attribute__((ext_vector_type(16))) __bf16       bf16x16;
typedef __attribute__((ext_vector_type(8)))  float        f32x8;
typedef __attribute__((ext_vector_type(4)))  unsigned int u32x4;
typedef __attribute__((ext_vector_type(8)))  int          i32x8;
typedef __attribute__((ext_vector_type(4)))  int          i32x4;

#if defined(__has_builtin)
#if __has_builtin(__builtin_amdgcn_tensor_load_to_lds) && \
    __has_builtin(__builtin_amdgcn_s_wait_tensorcnt)
#define HAVE_TDM 1
#endif
#endif
#ifndef HAVE_TDM
#define HAVE_TDM 0
#endif

union FragU { bf16x16 v; uint4 q[2]; };

// two runs of 8 bf16 (p, p+16) — B-fragment (lane = n, p at k = 8*half; e 0..7 -> k = 8*half+e, e 8..15 -> k = 16+8*half+(e-8))
__device__ __forceinline__ bf16x16 load_frag_contig(const __bf16* p) {
    FragU r;
    r.q[0] = *(const uint4*)(p);
    r.q[1] = *(const uint4*)(p + 16);
    return r.v;
}
// A-fragment: two runs of 8 (e 0..7 -> k = half*8+e ; e 8..15 -> k = 16+half*8+(e-8))
__device__ __forceinline__ bf16x16 load_frag_a(const __bf16* p0, const __bf16* p1) {
    FragU r;
    r.q[0] = *(const uint4*)(p0);
    r.q[1] = *(const uint4*)(p1);
    return r.v;
}

#define WMMA_BF16(A, B, C) \
    __builtin_amdgcn_wmma_f32_16x16x32_bf16(false, (A), false, (B), (short)0, (C), false, false)

static constexpr int T_SEQ = 2048;
static constexpr int DMODEL = 512;
static constexpr int NHEAD = 8;
static constexpr int HDIM = 64;

// ---------------------------------------------------------------------------
// Tensor Data Mover: 2D bf16 tile load -> LDS.  D# built per ISA cdna5 §8:
//  group0: [1:0]=count=1, [63:32]=lds_addr, [120:64]=global_addr, [127:126]=type=2
//  group1: [17:16]=data_size(1 => 2B), [79:48]=tensor_dim0, [111:80]=tensor_dim1,
//          [127:112]=tile_dim0, [143:128]=tile_dim1, [207:160]=tensor_dim0_stride
// Issued by one wave (EXEC ignored), completion via TENSORcnt.
// ---------------------------------------------------------------------------
#if HAVE_TDM
__device__ __forceinline__ void tdm_load_2d_to_lds(const __bf16* gsrc, uint32_t lds_off,
                                                   uint32_t dim0, uint32_t dim1,
                                                   uint32_t tile0, uint32_t tile1,
                                                   uint32_t stride0) {
    uint64_t ga = (uint64_t)(uintptr_t)gsrc;
    u32x4 g0;
    g0.x = 1u;                                                   // count=1 (user D#)
    g0.y = lds_off;                                              // lds_addr
    g0.z = (uint32_t)ga;                                         // global_addr[31:0]
    g0.w = (uint32_t)((ga >> 32) & 0x01FFFFFFu) | (2u << 30);    // addr[56:32] | type=2
    i32x8 g1;
    g1[0] = (int)(1u << 16);                                     // data_size=2B, mask=0
    g1[1] = (int)((dim0 & 0xFFFFu) << 16);                       // tensor_dim0[15:0]
    g1[2] = (int)((dim0 >> 16) | ((dim1 & 0xFFFFu) << 16));      // d0[31:16] | d1[15:0]
    g1[3] = (int)((dim1 >> 16) | (tile0 << 16));                 // d1[31:16] | tile_dim0
    g1[4] = (int)(tile1 & 0xFFFFu);                              // tile_dim1 (tile_dim2=0)
    g1[5] = (int)stride0;                                        // tensor_dim0_stride lo
    g1[6] = 0;
    g1[7] = 0;
    i32x4 z4 = {0, 0, 0, 0};
#if __clang_major__ >= 23
    i32x8 z8 = {0, 0, 0, 0, 0, 0, 0, 0};
    __builtin_amdgcn_tensor_load_to_lds(g0, g1, z4, z4, z8, 0);
#else
    __builtin_amdgcn_tensor_load_to_lds(g0, g1, z4, z4, 0);
#endif
}
#endif

// ---------------------------------------------------------------------------
// Prep kernels
// ---------------------------------------------------------------------------
__global__ __launch_bounds__(256) void cvt_bf16_kernel(const float* __restrict__ src,
                                                       __bf16* __restrict__ dst, int n) {
    int i = blockIdx.x * blockDim.x + threadIdx.x;
    if (i < n) dst[i] = (__bf16)src[i];
}

// src [K][N] fp32 -> dst [N][K] bf16 (B-fragments then load contiguously in k)
__global__ __launch_bounds__(256) void transpose_bf16_kernel(const float* __restrict__ src,
                                                             __bf16* __restrict__ dst,
                                                             int K, int N) {
    int i = blockIdx.x * blockDim.x + threadIdx.x;
    if (i < K * N) {
        int n = i / K, k = i - n * K;
        dst[i] = (__bf16)src[k * N + n];
    }
}

__global__ __launch_bounds__(256) void rope_table_kernel(float* __restrict__ cost,
                                                         float* __restrict__ sint) {
    int i = blockIdx.x * blockDim.x + threadIdx.x;
    if (i < T_SEQ * HDIM) {
        int t = i >> 6, d = i & 63;
        float inv_freq = powf(10000.0f, -((float)(d & 31) / 32.0f));
        float f = (float)t * inv_freq;
        cost[i] = cosf(f);
        sint[i] = sinf(f);
    }
}

// ---------------------------------------------------------------------------
// Fused QKV GEMM + bias + RoPE + scatter. Block: 128 thr = 4 waves, tile 64Mx64N.
// Weight tile (64 x 512 bf16 = 64 KB) staged in LDS once per block via TDM;
// A fragments register-pipelined across the K loop.
// ---------------------------------------------------------------------------
__global__ __launch_bounds__(128) void qkv_rope_kernel(
    const __bf16* __restrict__ xb,      // [M][512]
    const __bf16* __restrict__ wT,      // [1536][512]
    const float*  __restrict__ bqkv,    // [1536]
    const float*  __restrict__ cost,    // [T][64]
    const float*  __restrict__ sint,    // [T][64]
    __bf16* __restrict__ Qb,            // [B,H,T,64] (pre-scaled 1/sqrt(64))
    __bf16* __restrict__ Kb,            // [B,H,T,64]
    __bf16* __restrict__ Vt)            // [B,H,64,T]
{
    const int lane = threadIdx.x & 31;
    const int wave = threadIdx.x >> 5;
    const int l16  = lane & 15;
    const int half = lane >> 4;
    const int mbase = blockIdx.x * 64 + wave * 16;
    const int nbase = blockIdx.y * 64;

#if HAVE_TDM
    __shared__ __align__(16) __bf16 btile[64 * DMODEL];   // 64 KB weight tile
    if (wave == 0) {
        tdm_load_2d_to_lds(wT + (size_t)nbase * DMODEL,
                           (uint32_t)(uintptr_t)&btile[0],
                           DMODEL, 64, DMODEL, 64, DMODEL);
        __builtin_amdgcn_s_wait_tensorcnt(0);
    }
    __syncthreads();
#endif

    f32x8 acc[4] = {};
    const __bf16* arow = xb + (size_t)(mbase + l16) * DMODEL;

    bf16x16 af = load_frag_a(arow + half * 8, arow + 16 + half * 8);
    for (int kc = 0; kc < DMODEL; kc += 32) {
        bf16x16 afn = af;
        if (kc + 32 < DMODEL)   // prefetch next A chunk; overlaps with the 4 WMMAs
            afn = load_frag_a(arow + kc + 32 + half * 8, arow + kc + 48 + half * 8);
#pragma unroll
        for (int nt = 0; nt < 4; ++nt) {
#if HAVE_TDM
            const __bf16* bp = &btile[(size_t)(nt * 16 + l16) * DMODEL + kc + half * 8];
#else
            const __bf16* bp = wT + (size_t)(nbase + nt * 16 + l16) * DMODEL + kc + half * 8;
#endif
            bf16x16 bfr = load_frag_contig(bp);
            acc[nt] = WMMA_BF16(af, bfr, acc[nt]);
        }
        af = afn;
    }

    // ---- epilogue: bias, RoPE (q,k), scatter ----
    const int which = nbase >> 9;          // 0=q 1=k 2=v
    const int head  = (nbase & 511) >> 6;

    float res[4][8];
#pragma unroll
    for (int nt = 0; nt < 4; ++nt) {
        float bias = bqkv[nbase + nt * 16 + l16];
#pragma unroll
        for (int e = 0; e < 8; ++e) res[nt][e] = acc[nt][e] + bias;
    }

#pragma unroll
    for (int e = 0; e < 8; ++e) {
        int m = mbase + e + 8 * half;          // global row = b*T + t
        int b = m >> 11;
        int t = m & (T_SEQ - 1);
        int bh = b * NHEAD + head;
        if (which == 2) {                       // V: no RoPE, store transposed
#pragma unroll
            for (int nt = 0; nt < 4; ++nt) {
                int d = nt * 16 + l16;
                Vt[((size_t)bh * HDIM + d) * T_SEQ + t] = (__bf16)res[nt][e];
            }
        } else {                                // Q/K: RoPE fully in registers
            float rot[4];
#pragma unroll
            for (int nt = 0; nt < 4; ++nt) {
                int d = nt * 16 + l16;
                float c = cost[t * HDIM + d];
                float s = sint[t * HDIM + d];
                float partner = (nt < 2) ? -res[nt + 2][e] : res[nt - 2][e];
                rot[nt] = res[nt][e] * c + partner * s;
            }
            float scale = (which == 0) ? 0.125f : 1.0f;
            __bf16* dst = (which == 0) ? Qb : Kb;
#pragma unroll
            for (int nt = 0; nt < 4; ++nt) {
                int d = nt * 16 + l16;
                dst[((size_t)bh * T_SEQ + t) * HDIM + d] = (__bf16)(rot[nt] * scale);
            }
        }
    }
}

// ---------------------------------------------------------------------------
// Flash attention (causal). Grid: (B*H, T/64). Block: 128 = 4 waves.
// V-fragment loads issued before the softmax VALU block so their latency
// overlaps exp/shuffle work instead of stalling the P.V WMMAs.
// ---------------------------------------------------------------------------
__global__ __launch_bounds__(128) void flash_attn_kernel(
    const __bf16* __restrict__ Qb,   // [B,H,T,64] pre-scaled
    const __bf16* __restrict__ Kb,   // [B,H,T,64]
    const __bf16* __restrict__ Vt,   // [B,H,64,T]
    __bf16* __restrict__ Y)          // [B,T,512]
{
    __shared__ __align__(16) __bf16 pl[4][16][32];   // per-wave P staging (C->A layout)

    const int lane = threadIdx.x & 31;
    const int wave = threadIdx.x >> 5;
    const int l16  = lane & 15;
    const int half = lane >> 4;
    const int bh   = blockIdx.x;
    const int qrow0 = blockIdx.y * 64 + wave * 16;

    const __bf16* Qp = Qb + (size_t)bh * T_SEQ * HDIM;
    const __bf16* Kp = Kb + (size_t)bh * T_SEQ * HDIM;
    const __bf16* Vp = Vt + (size_t)bh * HDIM * T_SEQ;

    const __bf16* qr = Qp + (size_t)(qrow0 + l16) * HDIM;
    bf16x16 qf0 = load_frag_a(qr + half * 8,      qr + 16 + half * 8);
    bf16x16 qf1 = load_frag_a(qr + 32 + half * 8, qr + 48 + half * 8);

    f32x8 acc[4] = {};
    float mrow[8], lrow[8];
#pragma unroll
    for (int e = 0; e < 8; ++e) { mrow[e] = -3.0e38f; lrow[e] = 0.0f; }

    const f32x8 zero8 = {};

    for (int j = 0; j <= qrow0 + 15; j += 32) {
        // ---- issue all global loads for this block up front ----
        bf16x16 k00 = load_frag_contig(Kp + (size_t)(j      + l16) * HDIM + half * 8);
        bf16x16 k01 = load_frag_contig(Kp + (size_t)(j      + l16) * HDIM + 32 + half * 8);
        bf16x16 k10 = load_frag_contig(Kp + (size_t)(j + 16 + l16) * HDIM + half * 8);
        bf16x16 k11 = load_frag_contig(Kp + (size_t)(j + 16 + l16) * HDIM + 32 + half * 8);
        bf16x16 vf[4];
#pragma unroll
        for (int nt = 0; nt < 4; ++nt)
            vf[nt] = load_frag_contig(Vp + (size_t)(nt * 16 + l16) * T_SEQ + j + half * 8);

        // ---- scores: two 16x16 tiles (keys j..j+15, j+16..j+31) ----
        f32x8 s0 = WMMA_BF16(qf0, k00, zero8);
        s0       = WMMA_BF16(qf1, k01, s0);
        f32x8 s1 = WMMA_BF16(qf0, k10, zero8);
        s1       = WMMA_BF16(qf1, k11, s1);

        if (j + 31 > qrow0) {   // only boundary blocks need the causal mask
#pragma unroll
            for (int e = 0; e < 8; ++e) {
                int qrq = qrow0 + e + 8 * half;
                if (j      + l16 > qrq) s0[e] = -3.0e38f;
                if (j + 16 + l16 > qrq) s1[e] = -3.0e38f;
            }
        }

        // ---- online softmax (rows stripe across the 16-lane half) ----
#pragma unroll
        for (int e = 0; e < 8; ++e) {
            float a = s0[e], b = s1[e];
            float mx = fmaxf(a, b);
#pragma unroll
            for (int msk = 1; msk < 16; msk <<= 1) mx = fmaxf(mx, __shfl_xor(mx, msk, 32));
            float mnew  = fmaxf(mrow[e], mx);
            float p0    = __expf(a - mnew);
            float p1    = __expf(b - mnew);
            float alpha = __expf(mrow[e] - mnew);
            float rs = p0 + p1;
#pragma unroll
            for (int msk = 1; msk < 16; msk <<= 1) rs += __shfl_xor(rs, msk, 32);
            lrow[e] = lrow[e] * alpha + rs;
            mrow[e] = mnew;
#pragma unroll
            for (int nt = 0; nt < 4; ++nt) acc[nt][e] *= alpha;
            pl[wave][e + 8 * half][l16]      = (__bf16)p0;
            pl[wave][e + 8 * half][16 + l16] = (__bf16)p1;
        }

        // ---- P back in A-fragment layout (same wave; hw dscnt wait, no barrier) ----
        bf16x16 pf = load_frag_a(&pl[wave][l16][half * 8], &pl[wave][l16][16 + half * 8]);

        // ---- O += P.V ----
#pragma unroll
        for (int nt = 0; nt < 4; ++nt)
            acc[nt] = WMMA_BF16(pf, vf[nt], acc[nt]);
    }

    // ---- normalize and store Y [B,T,512] ----
    const int b = bh >> 3, h = bh & 7;
#pragma unroll
    for (int e = 0; e < 8; ++e) {
        float inv = 1.0f / lrow[e];
        int qrq = qrow0 + e + 8 * half;
        size_t rowoff = ((size_t)b * T_SEQ + qrq) * DMODEL + h * HDIM;
#pragma unroll
        for (int nt = 0; nt < 4; ++nt)
            Y[rowoff + nt * 16 + l16] = (__bf16)(acc[nt][e] * inv);
    }
}

// ---------------------------------------------------------------------------
// Output projection: out[M,512] = Y @ w_out + b_out (fp32). Same TDM staging.
// ---------------------------------------------------------------------------
__global__ __launch_bounds__(128) void out_proj_kernel(
    const __bf16* __restrict__ Yb,     // [M][512]
    const __bf16* __restrict__ wT,     // [512][512]
    const float*  __restrict__ bout,   // [512]
    float* __restrict__ out)           // [M][512] fp32
{
    const int lane = threadIdx.x & 31;
    const int wave = threadIdx.x >> 5;
    const int l16  = lane & 15;
    const int half = lane >> 4;
    const int mbase = blockIdx.x * 64 + wave * 16;
    const int nbase = blockIdx.y * 64;

#if HAVE_TDM
    __shared__ __align__(16) __bf16 btile[64 * DMODEL];
    if (wave == 0) {
        tdm_load_2d_to_lds(wT + (size_t)nbase * DMODEL,
                           (uint32_t)(uintptr_t)&btile[0],
                           DMODEL, 64, DMODEL, 64, DMODEL);
        __builtin_amdgcn_s_wait_tensorcnt(0);
    }
    __syncthreads();
#endif

    f32x8 acc[4] = {};
    const __bf16* arow = Yb + (size_t)(mbase + l16) * DMODEL;

    bf16x16 af = load_frag_a(arow + half * 8, arow + 16 + half * 8);
    for (int kc = 0; kc < DMODEL; kc += 32) {
        bf16x16 afn = af;
        if (kc + 32 < DMODEL)
            afn = load_frag_a(arow + kc + 32 + half * 8, arow + kc + 48 + half * 8);
#pragma unroll
        for (int nt = 0; nt < 4; ++nt) {
#if HAVE_TDM
            const __bf16* bp = &btile[(size_t)(nt * 16 + l16) * DMODEL + kc + half * 8];
#else
            const __bf16* bp = wT + (size_t)(nbase + nt * 16 + l16) * DMODEL + kc + half * 8;
#endif
            bf16x16 bfr = load_frag_contig(bp);
            acc[nt] = WMMA_BF16(af, bfr, acc[nt]);
        }
        af = afn;
    }

#pragma unroll
    for (int nt = 0; nt < 4; ++nt) {
        float bias = bout[nbase + nt * 16 + l16];
#pragma unroll
        for (int e = 0; e < 8; ++e) {
            int m = mbase + e + 8 * half;
            out[(size_t)m * DMODEL + nbase + nt * 16 + l16] = acc[nt][e] + bias;
        }
    }
}

// ---------------------------------------------------------------------------
// Host launcher
// ---------------------------------------------------------------------------
extern "C" void kernel_launch(void* const* d_in, const int* in_sizes, int n_in,
                              void* d_out, int out_size, void* d_ws, size_t ws_size,
                              hipStream_t stream) {
    const float* x     = (const float*)d_in[0];
    const float* w_qkv = (const float*)d_in[1];
    const float* b_qkv = (const float*)d_in[2];
    const float* w_out = (const float*)d_in[3];
    const float* b_out = (const float*)d_in[4];
    float* out = (float*)d_out;

    const int B = in_sizes[0] / (T_SEQ * DMODEL);
    const int M = B * T_SEQ;

    // ---- workspace layout ----
    char* p = (char*)d_ws;
    __bf16* xb    = (__bf16*)p;  p += (size_t)M * DMODEL * 2;
    __bf16* wqkvT = (__bf16*)p;  p += (size_t)3 * DMODEL * DMODEL * 2;
    __bf16* woutT = (__bf16*)p;  p += (size_t)DMODEL * DMODEL * 2;
    float*  cost  = (float*)p;   p += (size_t)T_SEQ * HDIM * 4;
    float*  sint  = (float*)p;   p += (size_t)T_SEQ * HDIM * 4;
    __bf16* Qb    = (__bf16*)p;  p += (size_t)M * DMODEL * 2;
    __bf16* Kb    = (__bf16*)p;  p += (size_t)M * DMODEL * 2;
    __bf16* Vt    = (__bf16*)p;  p += (size_t)M * DMODEL * 2;
    __bf16* Yb    = (__bf16*)p;  p += (size_t)M * DMODEL * 2;

    // ---- prep ----
    {
        int nx = M * DMODEL;
        cvt_bf16_kernel<<<(nx + 255) / 256, 256, 0, stream>>>(x, xb, nx);
        int nw1 = 3 * DMODEL * DMODEL;
        transpose_bf16_kernel<<<(nw1 + 255) / 256, 256, 0, stream>>>(w_qkv, wqkvT, DMODEL, 3 * DMODEL);
        int nw2 = DMODEL * DMODEL;
        transpose_bf16_kernel<<<(nw2 + 255) / 256, 256, 0, stream>>>(w_out, woutT, DMODEL, DMODEL);
        int nr = T_SEQ * HDIM;
        rope_table_kernel<<<(nr + 255) / 256, 256, 0, stream>>>(cost, sint);
    }

    // ---- fused QKV + RoPE ----
    qkv_rope_kernel<<<dim3(M / 64, (3 * DMODEL) / 64), 128, 0, stream>>>(
        xb, wqkvT, b_qkv, cost, sint, Qb, Kb, Vt);

    // ---- causal flash attention ----
    flash_attn_kernel<<<dim3(B * NHEAD, T_SEQ / 64), 128, 0, stream>>>(Qb, Kb, Vt, Yb);

    // ---- output projection ----
    out_proj_kernel<<<dim3(M / 64, DMODEL / 64), 128, 0, stream>>>(Yb, woutT, b_out, out);
}
